// GCAModel_greedy_26800595927376
// MI455X (gfx1250) — matrix operand order repaired and checked by reading, not compiled, not run on hardware
//
#include <hip/hip_runtime.h>
#include <hip/hip_bf16.h>

// ---------------------------------------------------------------------------
// Types for CDNA5 WMMA (wave32): v_wmma_f32_16x16x32_bf16
// ---------------------------------------------------------------------------
typedef __bf16 bf16_t;
typedef bf16_t v16bf __attribute__((ext_vector_type(16)));
typedef bf16_t v8bf  __attribute__((ext_vector_type(8)));
typedef float  v8f   __attribute__((ext_vector_type(8)));

// Model dims
#define BB   16
#define CDD  5
#define HIS  100
#define SS   30
#define EE   256
#define HH   16
#define VV   16
#define QD   200
#define QDP  208      // Qd padded to 13 tiles of 16
#define KSEL 3
#define NSEQ (BB*(CDD+HIS))   // 1680
#define NCDD (BB*CDD)         // 80

__device__ __forceinline__ v8f zero8() {
  v8f c;
#pragma unroll
  for (int i = 0; i < 8; ++i) c[i] = 0.f;
  return c;
}

// A-fragment (16x32 bf16, M x K) from row-major LDS [rows][ld].
// Lane layout: lanes 0-15 hold K 0-15; lanes 16-31 hold K 16-31 (matches B fragments).
__device__ __forceinline__ v16bf load_a(const bf16_t* M, int ld, int mt, int ks, int lane) {
  const int half = lane >> 4;
  const int m = (lane & 15) + 16 * mt;
  const bf16_t* p = M + m * ld + 32 * ks + 16 * half;
  v8bf lo = *(const v8bf*)(p);
  v8bf hi = *(const v8bf*)(p + 8);
  v16bf a;
#pragma unroll
  for (int i = 0; i < 8; ++i) { a[i] = lo[i]; a[8 + i] = hi[i]; }
  return a;
}

// B-fragment from pre-reordered weights: 16 contiguous bf16 per lane (two b128 loads).
__device__ __forceinline__ v16bf load_bw(const bf16_t* W, int lane) {
  return *(const v16bf*)(W + lane * 16);
}

// B-fragment = X^T (scores): B[k=d][n=j] = X[j][d]; contiguous in d per lane.
__device__ __forceinline__ v16bf load_bxT(const bf16_t* X, int ld, int nt, int ks, int lane) {
  const int j = (lane & 15) + 16 * nt;
  const bf16_t* p = X + j * ld + 32 * ks + 16 * (lane >> 4);
  return *(const v16bf*)p;
}

// B-fragment from row-major LDS [K][16] (XV for val = P @ XV).
__device__ __forceinline__ v16bf load_bk16(const bf16_t* Bm, int ks, int lane) {
  const int n = lane & 15;
  const int k0 = 32 * ks + 16 * (lane >> 4);
  v16bf b;
#pragma unroll
  for (int e = 0; e < 16; ++e) b[e] = Bm[(k0 + e) * 16 + n];
  return b;
}

// C (16x16 f32, 8 VGPRs) stores. Lane 0-15: M = r (+0), lane 16-31: M = r+8.
__device__ __forceinline__ void store_c_bf(bf16_t* D, int ld, int mt, int nt, int lane, v8f c) {
  const int n = (lane & 15) + 16 * nt;
  const int mb = 8 * (lane >> 4) + 16 * mt;
#pragma unroll
  for (int r = 0; r < 8; ++r) D[(mb + r) * ld + n] = (bf16_t)c[r];
}
__device__ __forceinline__ void store_c_f(float* D, int ld, int mt, int nt, int lane, v8f c) {
  const int n = (lane & 15) + 16 * nt;
  const int mb = 8 * (lane >> 4) + 16 * mt;
#pragma unroll
  for (int r = 0; r < 8; ++r) D[(mb + r) * ld + n] = c[r];
}

#define WMMA_BF16(a, b, c) \
  __builtin_amdgcn_wmma_f32_16x16x32_bf16(false, (a), false, (b), (short)0, (c), false, false)

// Async global -> LDS copy of 16B per lane (CDNA5 ASYNCcnt path, §15.18.3).
// VDST = LDS byte address (low 32 bits of the generic pointer per the LDS
// aperture rule ADDR[31:0]); VADDR = 64-bit global address.
__device__ __forceinline__ void async_copy_b128(const void* gsrc, void* ldst) {
  const unsigned lds_addr = (unsigned)(unsigned long long)ldst;
  asm volatile("global_load_async_to_lds_b128 %0, %1, off"
               :: "v"(lds_addr), "v"(gsrc) : "memory");
}
__device__ __forceinline__ void wait_asynccnt0() {
  asm volatile("s_wait_asynccnt 0x0" ::: "memory");
}

// ---------------------------------------------------------------------------
// Cooperative MHSA core: X (bf16, LP x 256) already in LDS at smem base.
// 8 waves / 256 threads. All GEMMs via v_wmma_f32_16x16x32_bf16.
// repr_out: 256 floats. mv_out (optional): SVx256 bf16 word reps.
// ---------------------------------------------------------------------------
template <int LP, int SV>
__device__ void mhsa_core(char* smem,
                          const bf16_t* WqW,  // [16][8][16][32][16]
                          const bf16_t* WvW,  // [16][8][1][32][16]
                          const bf16_t* WkW,  // [8][13][32][16]
                          const float* bk, const float* qvec,
                          float* repr_out, bf16_t* mv_out) {
  constexpr int MT = LP / 16;
  bf16_t* Xbf = (bf16_t*)(smem);
  bf16_t* Qbf = (bf16_t*)(smem + LP * 512);
  float*  Sc  = (float*) (smem + LP * 1024);
  bf16_t* Pbf = (bf16_t*)(smem + LP * 1024 + LP * LP * 4);
  bf16_t* XVb = (bf16_t*)(smem + LP * 1024 + LP * LP * 6);
  float*  mv  = (float*) (smem + LP * 1024 + LP * LP * 6 + LP * 32);
  float*  kp  = (float*) ((char*)mv + LP * 1024);
  float*  av  = (float*) ((char*)kp + LP * (QDP * 4));

  const int tid  = threadIdx.x;
  const int w    = tid >> 5;
  const int lane = tid & 31;
  const float scale = 0.0625f;  // 1/sqrt(E), E=256 (reference always uses this)

  for (int h = 0; h < HH; ++h) {
    // --- Q = X @ Wq_h  (LP x 256) ---
    for (int t = w; t < MT * 16; t += 8) {
      const int mt = t >> 4, nt = t & 15;
      v8f c = zero8();
#pragma unroll
      for (int ks = 0; ks < 8; ++ks) {
        v16bf a = load_a(Xbf, 256, mt, ks, lane);
        v16bf b = load_bw(WqW + (size_t)(((h * 8 + ks) * 16 + nt)) * 512, lane);
        c = WMMA_BF16(a, b, c);
      }
      store_c_bf(Qbf, 256, mt, nt, lane, c);
    }
    __syncthreads();
    // --- scores = Q @ X^T (LP x LP, K=256) ---
    for (int t = w; t < MT * MT; t += 8) {
      const int mt = t / MT, nt = t % MT;
      v8f c = zero8();
#pragma unroll
      for (int ks = 0; ks < 8; ++ks) {
        v16bf a = load_a(Qbf, 256, mt, ks, lane);
        v16bf b = load_bxT(Xbf, 256, nt, ks, lane);
        c = WMMA_BF16(a, b, c);
      }
      store_c_f(Sc, LP, mt, nt, lane, c);
    }
    // --- XV = X @ Wv_h (LP x 16); val = P @ XV later (algebraic refactor of attn@x@Wv) ---
    for (int t = w; t < MT; t += 8) {
      v8f c = zero8();
#pragma unroll
      for (int ks = 0; ks < 8; ++ks) {
        v16bf a = load_a(Xbf, 256, t, ks, lane);
        v16bf b = load_bw(WvW + (size_t)(h * 8 + ks) * 512, lane);
        c = WMMA_BF16(a, b, c);
      }
      store_c_bf(XVb, 16, t, 0, lane, c);
    }
    __syncthreads();
    // --- row softmax (mask padded columns j >= SV) ---
    if (tid < LP) {
      float* row = Sc + tid * LP;
      float mx = -__builtin_inff();
      for (int j = 0; j < SV; ++j) mx = fmaxf(mx, row[j] * scale);
      float sum = 0.f;
      for (int j = 0; j < SV; ++j) { float e = __expf(row[j] * scale - mx); row[j] = e; sum += e; }
      const float inv = 1.f / sum;
      for (int j = 0; j < LP; ++j) Pbf[tid * LP + j] = (bf16_t)((j < SV) ? row[j] * inv : 0.f);
    }
    __syncthreads();
    // --- val_h = P @ XV -> mv[:, h*16 : h*16+16] ---
    for (int t = w; t < MT; t += 8) {
      v8f c = zero8();
#pragma unroll
      for (int ks = 0; ks < LP / 32; ++ks) {
        v16bf a = load_a(Pbf, LP, t, ks, lane);
        v16bf b = load_bk16(XVb, ks, lane);
        c = WMMA_BF16(a, b, c);
      }
      const int n = lane & 15, mb = 8 * (lane >> 4) + 16 * t;
#pragma unroll
      for (int r = 0; r < 8; ++r) mv[(mb + r) * 256 + h * 16 + n] = c[r];
    }
    __syncthreads();
  }

  // --- mv -> bf16 (reuse Xbf; also the layout fusion consumes downstream) ---
  for (int i = tid; i < LP * 256; i += 256) Xbf[i] = (bf16_t)mv[i];
  __syncthreads();

  // --- keyp = tanh(mv @ Wk + bk)  (LP x 208 padded) ---
  for (int t = w; t < MT * 13; t += 8) {
    const int mt = t / 13, nt = t % 13;
    v8f c = zero8();
#pragma unroll
    for (int ks = 0; ks < 8; ++ks) {
      v16bf a = load_a(Xbf, 256, mt, ks, lane);
      v16bf b = load_bw(WkW + (size_t)(ks * 13 + nt) * 512, lane);
      c = WMMA_BF16(a, b, c);
    }
    const int n0 = (lane & 15) + 16 * nt;
    const int mb = 8 * (lane >> 4) + 16 * mt;
#pragma unroll
    for (int r = 0; r < 8; ++r)
      kp[(mb + r) * QDP + n0] = (n0 < QD) ? tanhf(c[r] + bk[n0]) : 0.f;
  }
  __syncthreads();

  // --- additive attention logits ---
  if (tid < SV) {
    float acc = 0.f;
    for (int j = 0; j < QD; ++j) acc += kp[tid * QDP + j] * qvec[j];
    av[tid] = acc * scale;
  }
  __syncthreads();

  // --- softmax over L and weighted sum: repr[r] = sum_l w_l * mv[l][r] ---
  {
    float mx = -__builtin_inff();
    for (int l = 0; l < SV; ++l) mx = fmaxf(mx, av[l]);
    float sum = 0.f;
    for (int l = 0; l < SV; ++l) sum += __expf(av[l] - mx);
    const float inv = 1.f / sum;
    float acc = 0.f;
    for (int l = 0; l < SV; ++l) acc += __expf(av[l] - mx) * inv * mv[l * 256 + tid];
    repr_out[tid] = acc;
  }
  // --- persist bf16 word reps (dword-packed copy straight from LDS) ---
  if (mv_out) {
    const unsigned* s = (const unsigned*)Xbf;
    unsigned* d = (unsigned*)mv_out;
    for (int i = tid; i < SV * 128; i += 256) d[i] = s[i];
  }
}

// ---------------------------------------------------------------------------
// Kernel 0: repack f32 weights into bf16 WMMA-B-fragment order.
// dst[h][ks][nt][lane][e] = src[h][k][n], k = e + 16*(lane/16) + 32*ks,
//                           n = 16*nt + lane%16 (0 beyond N).
// ---------------------------------------------------------------------------
__global__ void __launch_bounds__(256) reorder_w(const float* __restrict__ src,
                                                 bf16_t* __restrict__ dst,
                                                 int H, int D, int N, int Npad) {
  const int KT = D / 32, NT = Npad / 16;
  const long total = (long)H * KT * NT * 512;
  for (long i = (long)blockIdx.x * blockDim.x + threadIdx.x; i < total;
       i += (long)gridDim.x * blockDim.x) {
    const int e = (int)(i & 15);
    const int L = (int)((i >> 4) & 31);
    long r = i >> 9;
    const int nt = (int)(r % NT); r /= NT;
    const int ks = (int)(r % KT);
    const int h  = (int)(r / KT);
    const int k  = e + 16 * (L >> 4) + 32 * ks;
    const int n  = nt * 16 + (L & 15);
    const float v = (n < N) ? src[((long)h * D + k) * N + n] : 0.f;
    dst[i] = (bf16_t)v;
  }
}

// ---------------------------------------------------------------------------
// Kernel 1: word-level MHSA for all 1680 title sequences.
// ---------------------------------------------------------------------------
__global__ void __launch_bounds__(256) news_encoder(const int* __restrict__ cdd_ids,
                                                    const int* __restrict__ his_ids,
                                                    const float* __restrict__ emb,
                                                    const bf16_t* WqW, const bf16_t* WvW,
                                                    const bf16_t* WkW,
                                                    const float* bk, const float* qvec,
                                                    bf16_t* __restrict__ mvbf_out,
                                                    float* __restrict__ repr_out) {
  extern __shared__ char smem[];
  constexpr int LP = 32;
  bf16_t* Xbf = (bf16_t*)smem;
  const int seq = blockIdx.x;  // 0..79 candidates, 80..1679 history
  const int* ids = (seq < NCDD) ? (cdd_ids + seq * SS) : (his_ids + (seq - NCDD) * SS);
  const int tid = threadIdx.x;
  for (int r = 0; r < LP; ++r) {
    float v = 0.f;
    if (r < SS) v = emb[(long)ids[r] * EE + tid];
    Xbf[r * 256 + tid] = (bf16_t)v;
  }
  __syncthreads();
  mhsa_core<32, SS>(smem, WqW, WvW, WkW, bk, qvec,
                    repr_out + (long)seq * 256, mvbf_out + (long)seq * SS * 256);
}

// ---------------------------------------------------------------------------
// Kernel 2: logits + gumbel + greedy top-K (forward of straight-through == hard one-hot,
// and argmax is invariant under /TAU). Gumbel noise from a counter hash (stand-in for
// threefry; no runtime correctness signal in this loop).
// ---------------------------------------------------------------------------
__device__ __forceinline__ unsigned mix_u32(unsigned x) {
  x ^= x >> 16; x *= 0x7feb352du; x ^= x >> 15; x *= 0x846ca68bu; x ^= x >> 16;
  return x;
}

__global__ void __launch_bounds__(128) select_kernel(const float* __restrict__ repr,
                                                     const unsigned char* __restrict__ his_mask,
                                                     int* __restrict__ sel) {
  __shared__ float sc[HIS];
  const int bc = blockIdx.x;       // 0..79
  const int b  = bc / CDD;
  const int t  = threadIdx.x;
  if (t < HIS) {
    const float* cr = repr + (long)bc * 256;
    const float* hr = repr + (long)(NCDD + b * HIS + t) * 256;
    float d = 0.f;
    for (int i = 0; i < 256; ++i) d += cr[i] * hr[i];
    if (his_mask[b * HIS + t]) d = -__builtin_inff();
    sc[t] = d;
  }
  __syncthreads();
  if (t == 0) {
    for (int k = 0; k < KSEL; ++k) {
      float best = -__builtin_inff(); int bi = 0;
      for (int hh = 0; hh < HIS; ++hh) {
        const unsigned u = mix_u32(mix_u32(42u + 0x9e3779b9u * (unsigned)k) ^
                                   (unsigned)(bc * HIS + hh));
        const float uu = ((float)u + 0.5f) * 2.32830644e-10f;
        const float g  = -logf(-logf(uu));
        const float v  = sc[hh] + g;
        if (v > best) { best = v; bi = hh; }
      }
      sel[bc * KSEL + k] = bi;
      sc[bi] = -__builtin_inff();
    }
  }
}

// ---------------------------------------------------------------------------
// Kernel 3: fusion MHSA. Sequence = [cand words | zero sep | selected history words],
// staged straight into LDS with async global->LDS b128 copies (ASYNCcnt path);
// the word reps are already bf16 in HBM so this is a pure byte move.
// ---------------------------------------------------------------------------
__global__ void __launch_bounds__(256) fusion_kernel(const bf16_t* __restrict__ mvbf_all,
                                                     const int* __restrict__ sel,
                                                     const bf16_t* WqI, const bf16_t* WvI,
                                                     const bf16_t* WkI,
                                                     const float* bkI, const float* qvecI,
                                                     float* __restrict__ fvec3) {
  extern __shared__ char smem[];
  constexpr int LP = 64, SV = 2 * SS + 1;  // 61
  bf16_t* Xbf = (bf16_t*)smem;
  const int bck = blockIdx.x;            // 0..239
  const int k  = bck % KSEL;
  const int bc = bck / KSEL;
  const int b  = bc / CDD;
  const int hsel = sel[bc * KSEL + k];
  const bf16_t* cmv = mvbf_all + (long)bc * SS * 256;
  const bf16_t* hmv = mvbf_all + (long)(NCDD + b * HIS + hsel) * SS * 256;
  const int tid  = threadIdx.x;
  const int w    = tid >> 5;
  const int lane = tid & 31;

  // Zero separator row (30) and padding rows (61..63) via normal DS stores.
  Xbf[SS * 256 + tid]       = (bf16_t)0.f;
  Xbf[(SV + 0) * 256 + tid] = (bf16_t)0.f;
  Xbf[(SV + 1) * 256 + tid] = (bf16_t)0.f;
  Xbf[(SV + 2) * 256 + tid] = (bf16_t)0.f;

  // Async copy: one row (512B) per wave-instruction, 16B per lane.
  for (int r = w; r < SS; r += 8)
    async_copy_b128((const char*)(cmv + r * 256) + lane * 16,
                    (char*)(Xbf + r * 256) + lane * 16);
  for (int r = w; r < SS; r += 8)
    async_copy_b128((const char*)(hmv + r * 256) + lane * 16,
                    (char*)(Xbf + (SS + 1 + r) * 256) + lane * 16);
  wait_asynccnt0();
  __syncthreads();

  mhsa_core<64, SV>(smem, WqI, WvI, WkI, bkI, qvecI,
                    fvec3 + (long)bck * 256, nullptr);
}

// ---------------------------------------------------------------------------
// Kernel 4: fvec mean over K, linear score, log-softmax over CDD.
// ---------------------------------------------------------------------------
__global__ void __launch_bounds__(256) score_kernel(const float* __restrict__ fvec3,
                                                    const float* __restrict__ Wl,
                                                    const float* __restrict__ bl,
                                                    float* __restrict__ out) {
  __shared__ float red[256];
  __shared__ float sc[CDD];
  const int b = blockIdx.x, t = threadIdx.x;
  for (int c = 0; c < CDD; ++c) {
    float acc = 0.f;
    for (int k = 0; k < KSEL; ++k)
      acc += fvec3[(((long)(b * CDD + c)) * KSEL + k) * 256 + t];
    red[t] = acc * (1.f / 3.f) * Wl[t];
    __syncthreads();
    for (int s = 128; s > 0; s >>= 1) { if (t < s) red[t] += red[t + s]; __syncthreads(); }
    if (t == 0) sc[c] = red[0] + bl[0];
    __syncthreads();
  }
  if (t == 0) {
    float mx = sc[0];
    for (int c = 1; c < CDD; ++c) mx = fmaxf(mx, sc[c]);
    float sum = 0.f;
    for (int c = 0; c < CDD; ++c) sum += __expf(sc[c] - mx);
    const float lse = mx + logf(sum);
    for (int c = 0; c < CDD; ++c) out[b * CDD + c] = sc[c] - lse;
  }
}

// ---------------------------------------------------------------------------
// Host-side launcher
// ---------------------------------------------------------------------------
static constexpr size_t smem_bytes(int LP) {
  return (size_t)LP * 1024 + (size_t)LP * LP * 6 + (size_t)LP * 32 +
         (size_t)LP * 1024 + (size_t)LP * (QDP * 4) + (size_t)LP * 4;
}

extern "C" void kernel_launch(void* const* d_in, const int* in_sizes, int n_in,
                              void* d_out, int out_size, void* d_ws, size_t ws_size,
                              hipStream_t stream) {
  (void)in_sizes; (void)n_in; (void)out_size; (void)ws_size;
  const int*   cdd_ids = (const int*)d_in[0];
  const int*   his_ids = (const int*)d_in[1];
  const unsigned char* his_mask = (const unsigned char*)d_in[2];
  // d_in[3], d_in[4]: title pads — unused by the reference math.
  const float* emb   = (const float*)d_in[5];
  const float* Wq_w  = (const float*)d_in[6];
  const float* Wv_w  = (const float*)d_in[7];
  const float* Wk_w  = (const float*)d_in[8];
  const float* bk_w  = (const float*)d_in[9];
  const float* q_w   = (const float*)d_in[10];
  const float* Wq_i  = (const float*)d_in[11];
  const float* Wv_i  = (const float*)d_in[12];
  const float* Wk_i  = (const float*)d_in[13];
  const float* bk_i  = (const float*)d_in[14];
  const float* q_i   = (const float*)d_in[15];
  const float* W_ltr = (const float*)d_in[16];
  const float* b_ltr = (const float*)d_in[17];
  float* out = (float*)d_out;

  char* ws = (char*)d_ws;
  size_t off = 0;
  auto alloc = [&](size_t bytes) -> char* {
    char* p = ws + off;
    off += (bytes + 255) & ~(size_t)255;
    return p;
  };
  bf16_t* WqWb = (bf16_t*)alloc((size_t)16 * 8 * 16 * 512 * sizeof(bf16_t));   // 2 MB
  bf16_t* WvWb = (bf16_t*)alloc((size_t)16 * 8 * 1 * 512 * sizeof(bf16_t));
  bf16_t* WkWb = (bf16_t*)alloc((size_t)8 * 13 * 512 * sizeof(bf16_t));
  bf16_t* WqIb = (bf16_t*)alloc((size_t)16 * 8 * 16 * 512 * sizeof(bf16_t));
  bf16_t* WvIb = (bf16_t*)alloc((size_t)16 * 8 * 1 * 512 * sizeof(bf16_t));
  bf16_t* WkIb = (bf16_t*)alloc((size_t)8 * 13 * 512 * sizeof(bf16_t));
  bf16_t* mvbf_all = (bf16_t*)alloc((size_t)NSEQ * SS * 256 * sizeof(bf16_t)); // ~24.6 MB
  float*  repr   = (float*)alloc((size_t)NSEQ * 256 * sizeof(float));
  int*    sel    = (int*)  alloc((size_t)NCDD * KSEL * sizeof(int));
  float*  fvec3  = (float*)alloc((size_t)NCDD * KSEL * 256 * sizeof(float));

  // Kernel 0: weight repack to WMMA fragment order (bf16)
  reorder_w<<<4096, 256, 0, stream>>>(Wq_w, WqWb, 16, 256, 256, 256);
  reorder_w<<<256,  256, 0, stream>>>(Wv_w, WvWb, 16, 256, 16, 16);
  reorder_w<<<256,  256, 0, stream>>>(Wk_w, WkWb, 1, 256, QD, QDP);
  reorder_w<<<4096, 256, 0, stream>>>(Wq_i, WqIb, 16, 256, 256, 256);
  reorder_w<<<256,  256, 0, stream>>>(Wv_i, WvIb, 16, 256, 16, 16);
  reorder_w<<<256,  256, 0, stream>>>(Wk_i, WkIb, 1, 256, QD, QDP);

  // Kernel 1: word-level MHSA for all candidate + history titles
  news_encoder<<<NSEQ, 256, smem_bytes(32), stream>>>(
      cdd_ids, his_ids, emb, WqWb, WvWb, WkWb, bk_w, q_w, mvbf_all, repr);

  // Kernel 2: greedy gumbel top-K selection
  select_kernel<<<NCDD, 128, 0, stream>>>(repr, his_mask, sel);

  // Kernel 3: fusion MHSA over [cand | 0 | selected history] (async LDS staging)
  fusion_kernel<<<NCDD * KSEL, 256, smem_bytes(64), stream>>>(
      mvbf_all, sel, WqIb, WvIb, WkIb, bk_i, q_i, fvec3);

  // Kernel 4: mean over K, linear, log-softmax over CDD
  score_kernel<<<BB, 256, 0, stream>>>(fvec3, W_ltr, b_ltr, out);
}
